// RA_MLA_Attention_27092653703663
// MI455X (gfx1250) — matrix operand order repaired and checked by reading, not compiled, not run on hardware
//
#include <hip/hip_runtime.h>
#include <hip/hip_bf16.h>

// Model dims (fixed by the reference)
#define Tn 2048
#define En 2048
#define Hn 16
#define Ln 512
#define Dn 128
#define WIN 64

typedef __attribute__((ext_vector_type(16))) _Float16 v16h;
typedef __attribute__((ext_vector_type(8)))  _Float16 v8h;
typedef __attribute__((ext_vector_type(8)))  float    v8f;

union HFrag { v16h v; v8h h[2]; };

// A-operand (16x32, MxK): lanes 0-15 and 16-31 both hold rows M=0..15;
// element j -> K = (j<8 ? j : j+8) + 8*half  => two b128 loads at k0+8*half, +16.
__device__ __forceinline__ v16h load_a(const _Float16* __restrict__ base, int ld,
                                       int row, int k0, int half) {
  HFrag f;
  const _Float16* p = base + (size_t)row * ld + k0 + 8 * half;
  f.h[0] = *(const v8h*)(p);
  f.h[1] = *(const v8h*)(p + 16);
  return f.v;
}

// B-operand (32x16, KxN): lane holds column n=lane&15;
// element j -> K = (j<8 ? j : j+8) + 8*half  => two b128 loads at k0+8*g, +16.
__device__ __forceinline__ v16h load_b(const _Float16* __restrict__ base, int ld,
                                       int col, int k0, int g) {
  HFrag f;
  const _Float16* p = base + (size_t)col * ld + k0 + 8 * g;
  f.h[0] = *(const v8h*)(p);
  f.h[1] = *(const v8h*)(p + 16);
  return f.v;
}

#define WMMA16(AF, BF, CF) \
  __builtin_amdgcn_wmma_f32_16x16x32_f16(false, AF, false, BF, (short)0, CF, false, false)

// ---------------- fp32 -> f16 conversion ----------------
__global__ __launch_bounds__(256)
void k_cvt(const float* __restrict__ s, _Float16* __restrict__ d, int n) {
  int i = (blockIdx.x * 256 + threadIdx.x) * 4;
  if (i >= n) return;
  float4 f = *(const float4*)(s + i);
  d[i + 0] = (_Float16)f.x;
  d[i + 1] = (_Float16)f.y;
  d[i + 2] = (_Float16)f.z;
  d[i + 3] = (_Float16)f.w;
}

// [B][R][C] fp32 -> [B][C][R] f16 (transpose last two dims)
__global__ __launch_bounds__(256)
void k_tcvt(const float* __restrict__ s, _Float16* __restrict__ d, int R, int C) {
  int b = blockIdx.z;
  int c = blockIdx.x * 16 + (threadIdx.x & 15);
  int r = blockIdx.y * 16 + (threadIdx.x >> 4);
  d[(size_t)b * R * C + (size_t)c * R + r] =
      (_Float16)s[(size_t)b * R * C + (size_t)r * C + c];
}

// ------- batched WMMA GEMM (32x64 wave tile, ping-pong double-buffered K) ------
// C[M][N] = A[M][K] * Bcols, where Bcols[n][k] (row n = column n of B, K-contig).
// Requires K % 64 == 0 (true for K = 2048, 512, 128 here).
__global__ __launch_bounds__(256)
void k_gemm(const _Float16* __restrict__ A, int lda, long long sA,
            const _Float16* __restrict__ B, int ldb, long long sB,
            void* __restrict__ Cp, int ldc, long long sC, int cF32,
            int M, int N, int K, int batch) {
  int gw   = (int)((blockIdx.x * 256 + threadIdx.x) >> 5);
  int lane = threadIdx.x & 31;
  int half = lane >> 4, lx = lane & 15;
  int wN = N >> 6, wM = M >> 5;
  int perB = wM * wN;
  if (gw >= perB * batch) return;
  int b  = gw / perB, w = gw - b * perB;
  int mT = w / wN,   nG = w - mT * wN;
  const _Float16* Ab = A + (size_t)b * sA;
  const _Float16* Bb = B + (size_t)b * sB;

  v8f acc[2][4] = {};
  int r0 = mT * 32 + lx, r1 = r0 + 16;
  int cB = nG * 64 + lx;

  v16h xa0, xa1, xb[4], ya0, ya1, yb[4];

  // prologue: chunk 0 into X buffers
  xa0 = load_a(Ab, lda, r0, 0, half);
  xa1 = load_a(Ab, lda, r1, 0, half);
#pragma unroll
  for (int j = 0; j < 4; j++) xb[j] = load_b(Bb, ldb, cB + j * 16, 0, half);

  for (int k0 = 32; k0 < K; k0 += 64) {
    // load Y chunk (k0), compute X
    ya0 = load_a(Ab, lda, r0, k0, half);
    ya1 = load_a(Ab, lda, r1, k0, half);
#pragma unroll
    for (int j = 0; j < 4; j++) yb[j] = load_b(Bb, ldb, cB + j * 16, k0, half);
#pragma unroll
    for (int j = 0; j < 4; j++) {
      acc[0][j] = WMMA16(xa0, xb[j], acc[0][j]);
      acc[1][j] = WMMA16(xa1, xb[j], acc[1][j]);
    }
    // load X chunk (k0+32) if any, compute Y
    if (k0 + 32 < K) {
      xa0 = load_a(Ab, lda, r0, k0 + 32, half);
      xa1 = load_a(Ab, lda, r1, k0 + 32, half);
#pragma unroll
      for (int j = 0; j < 4; j++) xb[j] = load_b(Bb, ldb, cB + j * 16, k0 + 32, half);
    }
#pragma unroll
    for (int j = 0; j < 4; j++) {
      acc[0][j] = WMMA16(ya0, yb[j], acc[0][j]);
      acc[1][j] = WMMA16(ya1, yb[j], acc[1][j]);
    }
  }

  int col0 = nG * 64 + lx;
  if (cF32) {
    float* C = (float*)Cp + (size_t)b * sC;
#pragma unroll
    for (int u = 0; u < 2; u++) {
      int row0 = mT * 32 + u * 16 + 8 * half;
#pragma unroll
      for (int j = 0; j < 4; j++)
#pragma unroll
        for (int v = 0; v < 8; v++)
          C[(size_t)(row0 + v) * ldc + col0 + j * 16] = acc[u][j][v];
    }
  } else {
    _Float16* C = (_Float16*)Cp + (size_t)b * sC;
#pragma unroll
    for (int u = 0; u < 2; u++) {
      int row0 = mT * 32 + u * 16 + 8 * half;
#pragma unroll
      for (int j = 0; j < 4; j++)
#pragma unroll
        for (int v = 0; v < 8; v++)
          C[(size_t)(row0 + v) * ldc + col0 + j * 16] = (_Float16)acc[u][j][v];
    }
  }
}

// ---------------- flash attention (one wave per head x 16-query tile) ----------
// QL: [H][T][L] f16, LK: [T][L] f16, Vt: [H][D][T] f16, Ctx: [T][E] f16.
// V panel (128 d x 32 keys) is staged global->LDS with async copies overlapping
// the QK^T WMMA phase; consumed after s_wait_asynccnt.
__global__ __launch_bounds__(128)
void k_flash(const _Float16* __restrict__ QL,
             const _Float16* __restrict__ LK,
             const _Float16* __restrict__ Vt,
             _Float16* __restrict__ Ctx) {
  __shared__ _Float16 Ps[4][16 * 32];    // P staging (C-layout -> A-layout)
  __shared__ _Float16 Vs[4][128 * 32];   // async-staged V panel [d][k]
  int wib  = threadIdx.x >> 5;
  int wave = blockIdx.x * 4 + wib;
  int lane = threadIdx.x & 31;
  int half = lane >> 4, lx = lane & 15;
  int h = wave >> 7, qt = wave & 127;
  int qBase = qt << 4;
  const _Float16* QLh = QL + (size_t)h * Tn * Ln;
  const _Float16* Vth = Vt + (size_t)h * Dn * Tn;
  _Float16* P   = Ps[wib];
  _Float16* VsW = Vs[wib];
  // LDS aperture maps addr[31:0] directly to the LDS offset (ISA 10.2)
  unsigned vsBase = (unsigned)(unsigned long long)(void*)VsW;

  float rowm[8], rowl[8];
  v8f acc[8] = {};
#pragma unroll
  for (int v = 0; v < 8; v++) { rowm[v] = -__builtin_inff(); rowl[v] = 0.0f; }

  const float scale = 0.044194173824159216f;  // 1/sqrt(512)
  int d0 = lane >> 2, seg = lane & 3;         // per-lane async-copy geometry

  for (int sBase = 0; sBase < qBase + 16; sBase += 32) {
    // previous iteration's LDS reads must retire before we overwrite the panel
    asm volatile("s_wait_dscnt 0x0" ::: "memory");
    // kick async copy of V panel [128 x 32] into LDS (overlaps score WMMAs)
    {
      unsigned long long ga =
          (unsigned long long)(const void*)(Vth + (size_t)d0 * Tn + sBase) +
          (unsigned)seg * 16u;
      unsigned la = vsBase + (unsigned)lane * 16u;
#pragma unroll
      for (int t = 0; t < 16; t++) {
        asm volatile("global_load_async_to_lds_b128 %0, %1, off"
                     :: "v"(la), "v"(ga) : "memory");
        ga += (unsigned long long)(8 * Tn * 2);  // 8 d-rows per instruction step
        la += 512u;
      }
    }

    // scores: s = qlatent . latent_k^T over L (ping-pong double-buffered)
    v8f s0 = {}, s1 = {};
    v16h xa, xb0, xb1, ya, yb0, yb1;
    xa  = load_a(QLh, Ln, qBase + lx,      0, half);
    xb0 = load_b(LK,  Ln, sBase + lx,      0, half);
    xb1 = load_b(LK,  Ln, sBase + 16 + lx, 0, half);
    for (int l0 = 32; l0 < Ln; l0 += 64) {
      ya  = load_a(QLh, Ln, qBase + lx,      l0, half);
      yb0 = load_b(LK,  Ln, sBase + lx,      l0, half);
      yb1 = load_b(LK,  Ln, sBase + 16 + lx, l0, half);
      s0 = WMMA16(xa, xb0, s0);
      s1 = WMMA16(xa, xb1, s1);
      if (l0 + 32 < Ln) {
        xa  = load_a(QLh, Ln, qBase + lx,      l0 + 32, half);
        xb0 = load_b(LK,  Ln, sBase + lx,      l0 + 32, half);
        xb1 = load_b(LK,  Ln, sBase + 16 + lx, l0 + 32, half);
      }
      s0 = WMMA16(ya, yb0, s0);
      s1 = WMMA16(ya, yb1, s1);
    }

    // mask + reciprocal band boost + online softmax (C-layout registers)
    float alpha[8];
#pragma unroll
    for (int v = 0; v < 8; v++) {
      int trow = qBase + v + 8 * half;
      int c0 = sBase + lx, c1 = c0 + 16;
      float r0 = s0[v] * scale, r1 = s1[v] * scale;
      float x0 = (c0 > trow) ? -__builtin_inff() : ((trow - c0 < WIN) ? r0 * 1.5f : r0);
      float x1 = (c1 > trow) ? -__builtin_inff() : ((trow - c1 < WIN) ? r1 * 1.5f : r1);
      float m2 = fmaxf(x0, x1);
      m2 = fmaxf(m2, __shfl_xor(m2, 1, 32));
      m2 = fmaxf(m2, __shfl_xor(m2, 2, 32));
      m2 = fmaxf(m2, __shfl_xor(m2, 4, 32));
      m2 = fmaxf(m2, __shfl_xor(m2, 8, 32));
      float mnew = fmaxf(rowm[v], m2);
      float al = __expf(rowm[v] - mnew);
      rowm[v] = mnew;
      float p0 = __expf(x0 - mnew);
      float p1 = __expf(x1 - mnew);
      float ps = p0 + p1;
      ps += __shfl_xor(ps, 1, 32);
      ps += __shfl_xor(ps, 2, 32);
      ps += __shfl_xor(ps, 4, 32);
      ps += __shfl_xor(ps, 8, 32);
      rowl[v] = rowl[v] * al + ps;
      alpha[v] = al;
      int m = v + 8 * half;
      P[m * 32 + lx]      = (_Float16)p0;
      P[m * 32 + 16 + lx] = (_Float16)p1;
    }
    asm volatile("s_wait_dscnt 0x0" ::: "memory");
    HFrag pf;  // reload P in A-operand layout (16x32 over keys)
    {
      const _Float16* pp = P + lx * 32 + 8 * half;
      pf.h[0] = *(const v8h*)(pp);
      pf.h[1] = *(const v8h*)(pp + 16);
    }
#pragma unroll
    for (int j = 0; j < 8; j++)
#pragma unroll
      for (int v = 0; v < 8; v++) acc[j][v] *= alpha[v];

    // consume async-staged V panel
    asm volatile("s_wait_asynccnt 0x0" ::: "memory");
#pragma unroll
    for (int j = 0; j < 8; j++) {
      v16h bV = load_b(VsW, 32, j * 16 + lx, 0, half);
      acc[j] = WMMA16(pf.v, bV, acc[j]);
    }
  }
#pragma unroll
  for (int j = 0; j < 8; j++) {
    int col = h * Dn + j * 16 + lx;
#pragma unroll
    for (int v = 0; v < 8; v++) {
      int trow = qBase + v + 8 * half;
      Ctx[(size_t)trow * En + col] = (_Float16)(acc[j][v] / rowl[v]);
    }
  }
}

extern "C" void kernel_launch(void* const* d_in, const int* in_sizes, int n_in,
                              void* d_out, int out_size, void* d_ws, size_t ws_size,
                              hipStream_t stream) {
  const float* X   = (const float*)d_in[0];
  const float* Wq  = (const float*)d_in[1];
  const float* Wk  = (const float*)d_in[2];
  const float* Wv  = (const float*)d_in[3];
  const float* Qtl = (const float*)d_in[4];
  const float* Vup = (const float*)d_in[5];
  const float* Wo  = (const float*)d_in[6];
  float* Out = (float*)d_out;

  _Float16* p = (_Float16*)d_ws;
  auto take = [&](size_t n) { _Float16* r = p; p += n; return r; };
  _Float16* Xh   = take((size_t)Tn * En);
  _Float16* WqH  = take((size_t)En * En);
  _Float16* WkH  = take((size_t)Ln * En);
  _Float16* WvH  = take((size_t)Ln * En);
  _Float16* WoH  = take((size_t)En * En);
  _Float16* QtlT = take((size_t)Hn * Ln * Dn);   // [H][L][D]
  _Float16* VupT = take((size_t)Hn * Dn * Ln);   // [H][D][L]
  _Float16* Qh   = take((size_t)Tn * En);        // [T][H*D]
  _Float16* LKh  = take((size_t)Tn * Ln);        // [T][L]
  _Float16* LVh  = take((size_t)Tn * Ln);        // [T][L]
  _Float16* QLw  = take((size_t)Hn * Tn * Ln);   // [H][T][L]
  _Float16* Vtw  = take((size_t)Hn * Dn * Tn);   // [H][D][T]
  _Float16* CtxH = take((size_t)Tn * En);        // [T][H*D]

  auto cvt = [&](const float* s, _Float16* d, size_t n) {
    k_cvt<<<(int)((n / 4 + 255) / 256), 256, 0, stream>>>(s, d, (int)n);
  };
  cvt(X,  Xh,  (size_t)Tn * En);
  cvt(Wq, WqH, (size_t)En * En);
  cvt(Wk, WkH, (size_t)Ln * En);
  cvt(Wv, WvH, (size_t)Ln * En);
  cvt(Wo, WoH, (size_t)En * En);
  k_tcvt<<<dim3(Ln / 16, Dn / 16, Hn), 256, 0, stream>>>(Qtl, QtlT, Dn, Ln);
  k_tcvt<<<dim3(Dn / 16, Ln / 16, Hn), 256, 0, stream>>>(Vup, VupT, Ln, Dn);

  auto gemm = [&](const _Float16* A, int lda, long long sA,
                  const _Float16* B, int ldb, long long sB,
                  void* C, int ldc, long long sC, int cF32,
                  int M, int N, int K, int batch) {
    long long waves = (long long)(M / 32) * (N / 64) * batch;
    int blocks = (int)((waves + 7) / 8);
    k_gemm<<<blocks, 256, 0, stream>>>(A, lda, sA, B, ldb, sB, C, ldc, sC,
                                       cF32, M, N, K, batch);
  };

  // Q = X Wq^T            [T,E]
  gemm(Xh, En, 0, WqH, En, 0, Qh, En, 0, 0, Tn, En, En, 1);
  // latent_k = X Wk^T     [T,L]
  gemm(Xh, En, 0, WkH, En, 0, LKh, Ln, 0, 0, Tn, Ln, En, 1);
  // latent_v = X Wv^T     [T,L]
  gemm(Xh, En, 0, WvH, En, 0, LVh, Ln, 0, 0, Tn, Ln, En, 1);
  // q_latent[h] = Q[:,h,:] @ q_to_latent[h]   (batch over heads)
  gemm(Qh, En, Dn, QtlT, Dn, (long long)Ln * Dn,
       QLw, Ln, (long long)Tn * Ln, 0, Tn, Ln, Dn, Hn);
  // Vt[h] = v_up[h]^T @ latent_v^T  -> [D][T]  (batch over heads)
  gemm(VupT, Ln, (long long)Dn * Ln, LVh, Ln, 0,
       Vtw, Tn, (long long)Dn * Tn, 0, Dn, Tn, Ln, Hn);

  // attention: one wave per (head, 16-query tile) = 2048 waves, 4 waves/block
  k_flash<<<(Hn * (Tn / 16)) / 4, 128, 0, stream>>>(QLw, LKh, Vtw, CtxH);

  // out = ctx @ Wout^T  (fp32 output)
  gemm(CtxH, En, 0, WoH, En, 0, Out, En, 0, 1, Tn, En, En, 1);
}
